// Mamba2_66992899883183
// MI455X (gfx1250) — matrix operand order repaired and checked by reading, not compiled, not run on hardware
//
#include <hip/hip_runtime.h>

// ---------------- model dimensions ----------------
#define B_      4
#define L_      2048
#define EMB     256
#define NH      8
#define CHK     64
#define NTOK    (B_ * L_)           // 8192 tokens
#define DINNER  512
#define DSTATE  256
#define DCONV   4
#define CONVD   1024                // DINNER + 2*DSTATE
#define DPROJ   1544                // 2*DINNER + 2*DSTATE + NH
#define DPROJP  1600                // padded to 25*64 for 64-wide tiles
#define NC      (L_ / CHK)          // 32 chunks per sequence
#define EPSV    1e-5f

// ---------------- WMMA plumbing (CDNA5 wave32) ----------------
typedef __attribute__((ext_vector_type(16))) __bf16 v16bf;
typedef __attribute__((ext_vector_type(8)))  float  v8f;

union FragBF { v16bf v; __bf16 e[16]; unsigned short us[16]; unsigned int d[8]; };
union FragF  { v8f  v; float e[8]; };

__device__ __forceinline__ float  bf2f(__bf16 x) { return (float)x; }
__device__ __forceinline__ __bf16 f2bf(float x)  { return (__bf16)x; }

__device__ __forceinline__ v8f wmma_bf16f32(v16bf a, v16bf b, v8f c) {
  // D = A(16x32 bf16) * B(32x16 bf16) + C(16x16 f32)
  return __builtin_amdgcn_wmma_f32_16x16x32_bf16(false, a, false, b, (short)0, c, false, false);
}

// A-fragment as loaded here (k order matches B): lane's 16 elements are two contiguous
// 8-element runs:  A[m][16*half + 0..7]  and  A[m][16*half + 8..15].
// B-fragment (K-contiguous per column): elements i = B[16*half + i][n], i=0..15.
// => if a tile is stored K-contiguous for the needed row/column, every fragment
//    is exactly two 16-byte loads.

// async global->LDS copy of 16 bytes per lane (CDNA5 ASYNCcnt path)
#define ASYNC_G2L_B128(lds_ptr, gptr)                                          \
  do {                                                                         \
    unsigned __lds = (unsigned)(size_t)(lds_ptr);                              \
    asm volatile("global_load_async_to_lds_b128 %0, %1, off"                   \
                 :: "v"(__lds), "v"(gptr) : "memory");                         \
  } while (0)
#define WAIT_ASYNC() asm volatile("s_wait_asynccnt 0x0" ::: "memory")

// 256-thread (8-wave) block sum
__device__ __forceinline__ float block_reduce_sum(float v, float* red) {
  int tid = threadIdx.x;
  #pragma unroll
  for (int off = 16; off > 0; off >>= 1) v += __shfl_down(v, off, 32);
  if ((tid & 31) == 0) red[tid >> 5] = v;
  __syncthreads();
  if (tid == 0) {
    float t = 0.f;
    #pragma unroll
    for (int i = 0; i < 8; ++i) t += red[i];
    red[0] = t;
  }
  __syncthreads();
  return red[0];
}

// ---------------- embedding gather ----------------
__global__ __launch_bounds__(256) void embed_kernel(const float* __restrict__ x,
                                                    const float* __restrict__ emb,
                                                    float* __restrict__ h) {
  int tok = blockIdx.x;
  float xv = x[tok];
  int idx = (int)(255.0f * xv);           // trunc like astype(int32)
  idx = idx < 0 ? 0 : (idx > 255 ? 255 : idx);
  h[(size_t)tok * EMB + threadIdx.x] = emb[(size_t)idx * EMB + threadIdx.x];
}

// ---------------- rmsnorm (width 256) -> bf16 ----------------
__global__ __launch_bounds__(256) void rmsnorm256_bf16_kernel(const float* __restrict__ in,
                                                              const float* __restrict__ w,
                                                              __bf16* __restrict__ out) {
  __shared__ float red[8];
  int tok = blockIdx.x, tid = threadIdx.x;
  float v = in[(size_t)tok * EMB + tid];
  float ss = block_reduce_sum(v * v, red);
  float sc = rsqrtf(ss * (1.0f / EMB) + EPSV);
  out[(size_t)tok * EMB + tid] = f2bf(v * sc * w[tid]);
}

// ---------------- f32 -> bf16 weight convert with column padding ----------------
__global__ __launch_bounds__(256) void convert_pad_kernel(const float* __restrict__ src,
                                                          __bf16* __restrict__ dst,
                                                          int rows, int cs, int cd) {
  int idx = blockIdx.x * 256 + threadIdx.x;
  if (idx >= rows * cd) return;
  int r = idx / cd, c = idx % cd;
  dst[idx] = f2bf(c < cs ? src[(size_t)r * cs + c] : 0.0f);
}

// ---------------- generic bf16 WMMA GEMM: C[M,N] (+)= A[M,K] @ B[K,N] ----------------
// block tile 64x64, 8 waves; A tile staged via async-to-LDS; B tile staged transposed.
__global__ __launch_bounds__(256) void gemm_bf16_kernel(
    const __bf16* __restrict__ A, int lda,
    const __bf16* __restrict__ B, int ldb,
    float* __restrict__ C, int ldc,
    int K, int accumulate) {
  __shared__ __attribute__((aligned(32))) __bf16 Ash[64][40];   // [m][k], row pad for banks
  __shared__ __attribute__((aligned(32))) __bf16 Bsht[64][40];  // [n][k] transposed
  int tid = threadIdx.x;
  int wv = tid >> 5, lane = tid & 31, lhi = lane >> 4, l16 = lane & 15;
  int bm = blockIdx.y * 64, bn = blockIdx.x * 64;
  int tm = wv & 3, tn0 = (wv >> 2) * 2;
  FragF a0, a1;
  #pragma unroll
  for (int i = 0; i < 8; ++i) { a0.e[i] = 0.f; a1.e[i] = 0.f; }
  int ar = tid >> 2, ac = (tid & 3) * 8;   // A tile 64x32: 8 bf16 (16B) per thread
  int br = tid >> 3, bc = (tid & 7) * 8;   // B tile 32x64: 8 bf16 per thread
  int am = tm * 16 + l16;
  int n0 = tn0 * 16 + l16;
  for (int kb = 0; kb < K; kb += 32) {
    // A tile: async DMA straight into LDS (row-major copy, 16B per lane)
    ASYNC_G2L_B128(&Ash[ar][ac], A + (size_t)(bm + ar) * lda + kb + ac);
    // B tile: coalesced global load, transposed store into LDS
    {
      uint4 q = *(const uint4*)(B + (size_t)(kb + br) * ldb + bn + bc);
      const __bf16* e = (const __bf16*)&q;
      #pragma unroll
      for (int i = 0; i < 8; ++i) Bsht[bc + i][br] = e[i];
    }
    if (kb + 32 < K)
      __builtin_prefetch(A + (size_t)(bm + ar) * lda + kb + 32 + ac, 0, 1);
    WAIT_ASYNC();
    __syncthreads();
    FragBF af, b0, b1;
    *(uint4*)&af.us[0] = *(const uint4*)&Ash[am][16 * lhi];
    *(uint4*)&af.us[8] = *(const uint4*)&Ash[am][16 * lhi + 8];
    *(uint4*)&b0.us[0] = *(const uint4*)&Bsht[n0][16 * lhi];
    *(uint4*)&b0.us[8] = *(const uint4*)&Bsht[n0][16 * lhi + 8];
    *(uint4*)&b1.us[0] = *(const uint4*)&Bsht[n0 + 16][16 * lhi];
    *(uint4*)&b1.us[8] = *(const uint4*)&Bsht[n0 + 16][16 * lhi + 8];
    a0.v = wmma_bf16f32(af.v, b0.v, a0.v);
    a1.v = wmma_bf16f32(af.v, b1.v, a1.v);
    __syncthreads();
  }
  #pragma unroll
  for (int r = 0; r < 8; ++r) {
    int m = bm + tm * 16 + lhi * 8 + r;
    int nn = bn + tn0 * 16 + l16;
    float* p0 = C + (size_t)m * ldc + nn;
    float* p1 = p0 + 16;
    if (accumulate) { *p0 += a0.e[r]; *p1 += a1.e[r]; }
    else            { *p0 = a0.e[r];  *p1 = a1.e[r]; }
  }
}

// ---------------- dt = softplus(raw + bias); acs = cumsum(A*dt) within chunk ----------------
__global__ __launch_bounds__(64) void dt_cumsum_kernel(
    const float* __restrict__ zx, const float* __restrict__ dt_bias,
    const float* __restrict__ A_log, float* __restrict__ dt_out,
    float* __restrict__ acs) {
  int blk = blockIdx.x;
  int c = blk % NC, hh = (blk / NC) % NH, b = blk / (NC * NH);
  int l = threadIdx.x;
  int gtok = b * L_ + c * CHK + l;
  float raw = zx[(size_t)gtok * DPROJP + (DPROJ - NH) + hh] + dt_bias[hh];
  float dtv = raw > 20.0f ? raw : log1pf(__expf(raw));
  dt_out[gtok * NH + hh] = dtv;
  float a = -__expf(A_log[hh]) * dtv;
  __shared__ float sc[CHK];
  sc[l] = a;
  __syncthreads();
  for (int off = 1; off < CHK; off <<= 1) {
    float add = (l >= off) ? sc[l - off] : 0.0f;
    __syncthreads();
    sc[l] += add;
    __syncthreads();
  }
  acs[(((size_t)(b * NH + hh) * NC) + c) * CHK + l] = sc[l];
}

// ---------------- depthwise causal conv(4) + silu; split into xh / x*dt / B / C ----------------
__global__ __launch_bounds__(256) void conv_silu_kernel(
    const float* __restrict__ zx, const float* __restrict__ cw,
    const float* __restrict__ cb, const float* __restrict__ dt,
    float* __restrict__ xh, __bf16* __restrict__ xdt,
    __bf16* __restrict__ Bbf, __bf16* __restrict__ Cbf) {
  int gtok = blockIdx.x;
  int b = gtok / L_, t = gtok % L_;
  for (int ch = threadIdx.x; ch < CONVD; ch += 256) {
    float acc = cb[ch];
    #pragma unroll
    for (int k = 0; k < DCONV; ++k) {
      int tt = t - (DCONV - 1) + k;
      if (tt >= 0)
        acc += zx[(size_t)(b * L_ + tt) * DPROJP + DINNER + ch] * cw[ch * DCONV + k];
    }
    float s = acc / (1.0f + __expf(-acc));    // silu
    if (ch < DINNER) {
      xh[(size_t)gtok * DINNER + ch] = s;
      xdt[(size_t)gtok * DINNER + ch] = f2bf(s * dt[gtok * NH + (ch >> 6)]);
    } else if (ch < DINNER + DSTATE) {
      Bbf[(size_t)gtok * DSTATE + (ch - DINNER)] = f2bf(s);
    } else {
      Cbf[(size_t)gtok * DSTATE + (ch - DINNER - DSTATE)] = f2bf(s);
    }
  }
}

// ---------------- SSD intra-chunk: Y = ((C@B^T) * mask*exp(acs_l-acs_s)) @ x + D*xh ----------------
__global__ __launch_bounds__(256) void ssd_diag_kernel(
    const __bf16* __restrict__ Cbf, const __bf16* __restrict__ Bbf,
    const __bf16* __restrict__ xdt, const float* __restrict__ xh,
    const float* __restrict__ acs, const float* __restrict__ Dvec,
    float* __restrict__ Y) {
  __shared__ float acs_sh[CHK];
  __shared__ __attribute__((aligned(32))) __bf16 Gsh[CHK][72];   // [l][s], padded
  __shared__ __attribute__((aligned(32))) __bf16 xsht[CHK][72];  // [p][s] transposed
  int blk = blockIdx.x;
  int hh = blk % NH, c = (blk / NH) % NC, b = blk / (NH * NC);
  int tb = b * L_ + c * CHK;
  int tid = threadIdx.x;
  int wv = tid >> 5, lane = tid & 31, lhi = lane >> 4, l16 = lane & 15;
  int tm = wv & 3, tn0 = (wv >> 2) * 2;
  if (tid < CHK) acs_sh[tid] = acs[(((size_t)(b * NH + hh) * NC) + c) * CHK + tid];
  { // stage x chunk (64 s x 64 p, this head) transposed: xsht[p][s]
    int s = tid >> 2, pb = (tid & 3) * 16;
    const __bf16* src = xdt + (size_t)(tb + s) * DINNER + hh * 64 + pb;
    uint4 q0 = *(const uint4*)src;
    uint4 q1 = *(const uint4*)(src + 8);
    const __bf16* e0 = (const __bf16*)&q0;
    const __bf16* e1 = (const __bf16*)&q1;
    #pragma unroll
    for (int i = 0; i < 8; ++i) xsht[pb + i][s] = e0[i];
    #pragma unroll
    for (int i = 0; i < 8; ++i) xsht[pb + 8 + i][s] = e1[i];
  }
  __syncthreads();

  // Phase 1: G = C @ B^T  (M=l, N=s, K=n=256), operands straight from global
  FragF g0, g1;
  #pragma unroll
  for (int i = 0; i < 8; ++i) { g0.e[i] = 0.f; g1.e[i] = 0.f; }
  {
    const __bf16* arow  = Cbf + (size_t)(tb + tm * 16 + l16) * DSTATE;
    const __bf16* brow0 = Bbf + (size_t)(tb + tn0 * 16 + l16) * DSTATE;
    const __bf16* brow1 = Bbf + (size_t)(tb + tn0 * 16 + 16 + l16) * DSTATE;
    for (int kb = 0; kb < DSTATE; kb += 32) {
      FragBF af, b0, b1;
      *(uint4*)&af.us[0] = *(const uint4*)(arow + kb + 16 * lhi);
      *(uint4*)&af.us[8] = *(const uint4*)(arow + kb + 16 * lhi + 8);
      *(uint4*)&b0.us[0] = *(const uint4*)(brow0 + kb + 16 * lhi);
      *(uint4*)&b0.us[8] = *(const uint4*)(brow0 + kb + 16 * lhi + 8);
      *(uint4*)&b1.us[0] = *(const uint4*)(brow1 + kb + 16 * lhi);
      *(uint4*)&b1.us[8] = *(const uint4*)(brow1 + kb + 16 * lhi + 8);
      g0.v = wmma_bf16f32(af.v, b0.v, g0.v);
      g1.v = wmma_bf16f32(af.v, b1.v, g1.v);
    }
  }
  // Phase 2: causal mask + segment decay, park as bf16 in LDS
  #pragma unroll
  for (int r = 0; r < 8; ++r) {
    int l = tm * 16 + lhi * 8 + r;
    int s0 = tn0 * 16 + l16, s1 = s0 + 16;
    float al = acs_sh[l];
    Gsh[l][s0] = f2bf(s0 <= l ? g0.e[r] * __expf(al - acs_sh[s0]) : 0.0f);
    Gsh[l][s1] = f2bf(s1 <= l ? g1.e[r] * __expf(al - acs_sh[s1]) : 0.0f);
  }
  __syncthreads();
  // Phase 3: Y_diag = G @ x  (K = s = 64)
  FragF y0, y1;
  #pragma unroll
  for (int i = 0; i < 8; ++i) { y0.e[i] = 0.f; y1.e[i] = 0.f; }
  for (int kb = 0; kb < CHK; kb += 32) {
    FragBF af, b0, b1;
    *(uint4*)&af.us[0] = *(const uint4*)&Gsh[tm * 16 + l16][kb + 16 * lhi];
    *(uint4*)&af.us[8] = *(const uint4*)&Gsh[tm * 16 + l16][kb + 16 * lhi + 8];
    *(uint4*)&b0.us[0] = *(const uint4*)&xsht[tn0 * 16 + l16][kb + 16 * lhi];
    *(uint4*)&b0.us[8] = *(const uint4*)&xsht[tn0 * 16 + l16][kb + 16 * lhi + 8];
    *(uint4*)&b1.us[0] = *(const uint4*)&xsht[tn0 * 16 + 16 + l16][kb + 16 * lhi];
    *(uint4*)&b1.us[8] = *(const uint4*)&xsht[tn0 * 16 + 16 + l16][kb + 16 * lhi + 8];
    y0.v = wmma_bf16f32(af.v, b0.v, y0.v);
    y1.v = wmma_bf16f32(af.v, b1.v, y1.v);
  }
  float Dh = Dvec[hh];
  #pragma unroll
  for (int r = 0; r < 8; ++r) {
    int l = tm * 16 + lhi * 8 + r;
    int p0 = tn0 * 16 + l16;
    size_t base = (size_t)(tb + l) * DINNER + hh * 64;
    Y[base + p0]      = y0.e[r] + Dh * xh[base + p0];
    Y[base + p0 + 16] = y1.e[r] + Dh * xh[base + p0 + 16];
  }
}

// ---------------- SSD inter-chunk scan: running state S[64p x 256n] lives in LDS ----------------
// one block per (b, head), sequential over 32 chunks:
//   Y += exp(acs_l) * C @ S^T ;  S = exp(acs_last)*S + x^T @ (B * decay)
__global__ __launch_bounds__(256) void ssd_state_kernel(
    const __bf16* __restrict__ Cbf, const __bf16* __restrict__ Bbf,
    const __bf16* __restrict__ xdt, const float* __restrict__ acs,
    float* __restrict__ Y) {
  __shared__ __attribute__((aligned(32))) float  S[64][DSTATE];     // 64 KB master (f32)
  __shared__ __attribute__((aligned(32))) __bf16 Sbf[64][DSTATE];   // 32 KB bf16 shadow
  __shared__ __attribute__((aligned(32))) __bf16 xsht[CHK][72];     // [p][l] transposed
  __shared__ __attribute__((aligned(32))) __bf16 scBt[DSTATE][CHK]; // [n][l] (B*decay)^T
  __shared__ float acs_sh[CHK];
  __shared__ float dec_sh[CHK];
  int blk = blockIdx.x;
  int hh = blk % NH, b = blk / NH;
  int tid = threadIdx.x;
  int wv = tid >> 5, lane = tid & 31, lhi = lane >> 4, l16 = lane & 15;
  int tm = wv & 3, tn0 = (wv >> 2) * 2;
  for (int i = tid; i < 64 * DSTATE; i += 256) {
    ((float*)S)[i] = 0.0f;
    ((__bf16*)Sbf)[i] = f2bf(0.0f);
  }
  __syncthreads();
  for (int c = 0; c < NC; ++c) {
    int tb = b * L_ + c * CHK;
    if (tid < CHK) acs_sh[tid] = acs[(((size_t)(b * NH + hh) * NC) + c) * CHK + tid];
    __syncthreads();
    float alast = acs_sh[CHK - 1];
    if (tid < CHK) dec_sh[tid] = __expf(alast - acs_sh[tid]);

    // Phase A: Yoff = C @ S^T  (M=l, N=p, K=n=256); S read from bf16 shadow in LDS
    FragF y0, y1;
    #pragma unroll
    for (int i = 0; i < 8; ++i) { y0.e[i] = 0.f; y1.e[i] = 0.f; }
    const __bf16* arow = Cbf + (size_t)(tb + tm * 16 + l16) * DSTATE;
    int p0 = tn0 * 16 + l16;
    for (int kb = 0; kb < DSTATE; kb += 32) {
      FragBF af, b0, b1;
      *(uint4*)&af.us[0] = *(const uint4*)(arow + kb + 16 * lhi);
      *(uint4*)&af.us[8] = *(const uint4*)(arow + kb + 16 * lhi + 8);
      *(uint4*)&b0.us[0] = *(const uint4*)&Sbf[p0][kb + 16 * lhi];
      *(uint4*)&b0.us[8] = *(const uint4*)&Sbf[p0][kb + 16 * lhi + 8];
      *(uint4*)&b1.us[0] = *(const uint4*)&Sbf[p0 + 16][kb + 16 * lhi];
      *(uint4*)&b1.us[8] = *(const uint4*)&Sbf[p0 + 16][kb + 16 * lhi + 8];
      y0.v = wmma_bf16f32(af.v, b0.v, y0.v);
      y1.v = wmma_bf16f32(af.v, b1.v, y1.v);
    }
    #pragma unroll
    for (int r = 0; r < 8; ++r) {
      int l = tm * 16 + lhi * 8 + r;
      float el = __expf(acs_sh[l]);
      size_t base = (size_t)(tb + l) * DINNER + hh * 64;
      Y[base + p0]      += el * y0.e[r];
      Y[base + p0 + 16] += el * y1.e[r];
    }
    __syncthreads();   // all Sbf reads done; dec_sh visible

    // stage x chunk transposed and (B*decay)^T
    {
      int s = tid >> 2, pb = (tid & 3) * 16;
      const __bf16* src = xdt + (size_t)(tb + s) * DINNER + hh * 64 + pb;
      uint4 q0 = *(const uint4*)src;
      uint4 q1 = *(const uint4*)(src + 8);
      const __bf16* e0 = (const __bf16*)&q0;
      const __bf16* e1 = (const __bf16*)&q1;
      #pragma unroll
      for (int i = 0; i < 8; ++i) xsht[pb + i][s] = e0[i];
      #pragma unroll
      for (int i = 0; i < 8; ++i) xsht[pb + 8 + i][s] = e1[i];
    }
    {
      int l = tid >> 2, nb = (tid & 3) * 64;
      float d = dec_sh[l];
      const __bf16* src = Bbf + (size_t)(tb + l) * DSTATE + nb;
      #pragma unroll 8
      for (int i = 0; i < 64; ++i) scBt[nb + i][l] = f2bf(bf2f(src[i]) * d);
    }
    __syncthreads();

    // Phase B: dS = x^T @ scB  (M=p=64, N=n=256, K=l=64); 8 n-tiles per wave
    int tp = wv & 3;
    int tnb = (wv >> 2) * 8;
    FragF sacc[8];
    #pragma unroll
    for (int t8 = 0; t8 < 8; ++t8)
      #pragma unroll
      for (int i = 0; i < 8; ++i) sacc[t8].e[i] = 0.f;
    int pl = tp * 16 + l16;
    for (int kb = 0; kb < CHK; kb += 32) {
      FragBF af;
      *(uint4*)&af.us[0] = *(const uint4*)&xsht[pl][kb + 16 * lhi];
      *(uint4*)&af.us[8] = *(const uint4*)&xsht[pl][kb + 16 * lhi + 8];
      #pragma unroll
      for (int tn = 0; tn < 8; ++tn) {
        FragBF bv;
        int ncol = (tnb + tn) * 16 + l16;
        *(uint4*)&bv.us[0] = *(const uint4*)&scBt[ncol][kb + 16 * lhi];
        *(uint4*)&bv.us[8] = *(const uint4*)&scBt[ncol][kb + 16 * lhi + 8];
        sacc[tn].v = wmma_bf16f32(af.v, bv.v, sacc[tn].v);
      }
    }
    float cdec = __expf(alast);
    #pragma unroll
    for (int tn = 0; tn < 8; ++tn)
      #pragma unroll
      for (int r = 0; r < 8; ++r) {
        int p = tp * 16 + lhi * 8 + r;
        int n = (tnb + tn) * 16 + l16;
        float sv = S[p][n] * cdec + sacc[tn].e[r];
        S[p][n] = sv;
        Sbf[p][n] = f2bf(sv);
      }
    __syncthreads();
  }
}

// ---------------- gated rmsnorm: out = rmsnorm(Y * silu(z)) * gw, width 512 ----------------
__global__ __launch_bounds__(256) void gate_rmsnorm_kernel(
    const float* __restrict__ Y, const float* __restrict__ zx,
    const float* __restrict__ gw, __bf16* __restrict__ out) {
  __shared__ float red[8];
  int tok = blockIdx.x, tid = threadIdx.x;
  float vals[2];
  float ss = 0.f;
  #pragma unroll
  for (int e = 0; e < 2; ++e) {
    int j = tid + e * 256;
    float z = zx[(size_t)tok * DPROJP + j];
    float v = Y[(size_t)tok * DINNER + j] * (z / (1.0f + __expf(-z)));
    vals[e] = v;
    ss += v * v;
  }
  float tot = block_reduce_sum(ss, red);
  float sc = rsqrtf(tot * (1.0f / DINNER) + EPSV);
  #pragma unroll
  for (int e = 0; e < 2; ++e) {
    int j = tid + e * 256;
    out[(size_t)tok * DINNER + j] = f2bf(vals[e] * sc * gw[j]);
  }
}

// ---------------- host orchestration ----------------
extern "C" void kernel_launch(void* const* d_in, const int* in_sizes, int n_in,
                              void* d_out, int out_size, void* d_ws, size_t ws_size,
                              hipStream_t stream) {
  (void)in_sizes; (void)n_in; (void)out_size; (void)ws_size;
  const float* x        = (const float*)d_in[0];
  const float* emb      = (const float*)d_in[1];
  const float* in_projw = (const float*)d_in[2];
  const float* conv_w   = (const float*)d_in[3];
  const float* conv_b   = (const float*)d_in[4];
  const float* dt_bias  = (const float*)d_in[5];
  const float* A_log    = (const float*)d_in[6];
  const float* Dvec     = (const float*)d_in[7];
  const float* gnorm_w  = (const float*)d_in[8];
  const float* outp_w   = (const float*)d_in[9];
  const float* norm_w   = (const float*)d_in[10];
  const float* normf_w  = (const float*)d_in[11];
  const float* head_w   = (const float*)d_in[12];

  char* ws = (char*)d_ws;
  size_t off = 0;
  auto alloc = [&](size_t bytes) -> void* {
    void* p = ws + off;
    off += (bytes + 255) & ~(size_t)255;
    return p;
  };
  float*  h    = (float*) alloc((size_t)NTOK * EMB * 4);
  __bf16* ubf  = (__bf16*)alloc((size_t)NTOK * EMB * 2);
  __bf16* Wbf  = (__bf16*)alloc((size_t)DINNER * DPROJP * 2);  // reused for all weights
  float*  zx   = (float*) alloc((size_t)NTOK * DPROJP * 4);
  float*  dt   = (float*) alloc((size_t)NTOK * NH * 4);
  float*  acs  = (float*) alloc((size_t)B_ * NH * NC * CHK * 4);
  float*  xh   = (float*) alloc((size_t)NTOK * DINNER * 4);
  __bf16* xdt  = (__bf16*)alloc((size_t)NTOK * DINNER * 2);
  __bf16* Bbf  = (__bf16*)alloc((size_t)NTOK * DSTATE * 2);
  __bf16* Cbf  = (__bf16*)alloc((size_t)NTOK * DSTATE * 2);
  float*  Yb   = (float*) alloc((size_t)NTOK * DINNER * 4);
  __bf16* ybf  = (__bf16*)alloc((size_t)NTOK * DINNER * 2);
  __bf16* hbf  = (__bf16*)alloc((size_t)NTOK * EMB * 2);

  embed_kernel<<<NTOK, 256, 0, stream>>>(x, emb, h);

  for (int layer = 0; layer < 2; ++layer) {
    rmsnorm256_bf16_kernel<<<NTOK, 256, 0, stream>>>(h, norm_w + layer * EMB, ubf);
    convert_pad_kernel<<<(EMB * DPROJP + 255) / 256, 256, 0, stream>>>(
        in_projw + (size_t)layer * EMB * DPROJ, Wbf, EMB, DPROJ, DPROJP);
    dim3 g1(DPROJP / 64, NTOK / 64);
    gemm_bf16_kernel<<<g1, 256, 0, stream>>>(ubf, EMB, Wbf, DPROJP, zx, DPROJP, EMB, 0);

    dt_cumsum_kernel<<<B_ * NH * NC, 64, 0, stream>>>(
        zx, dt_bias + layer * NH, A_log + layer * NH, dt, acs);
    conv_silu_kernel<<<NTOK, 256, 0, stream>>>(
        zx, conv_w + (size_t)layer * CONVD * DCONV, conv_b + (size_t)layer * CONVD,
        dt, xh, xdt, Bbf, Cbf);

    ssd_diag_kernel<<<B_ * NC * NH, 256, 0, stream>>>(
        Cbf, Bbf, xdt, xh, acs, Dvec + layer * NH, Yb);
    ssd_state_kernel<<<B_ * NH, 256, 0, stream>>>(Cbf, Bbf, xdt, acs, Yb);

    gate_rmsnorm_kernel<<<NTOK, 256, 0, stream>>>(Yb, zx, gnorm_w + (size_t)layer * DINNER, ybf);
    convert_pad_kernel<<<(DINNER * EMB + 255) / 256, 256, 0, stream>>>(
        outp_w + (size_t)layer * DINNER * EMB, Wbf, DINNER, EMB, EMB);
    dim3 g2(EMB / 64, NTOK / 64);
    gemm_bf16_kernel<<<g2, 256, 0, stream>>>(ybf, DINNER, Wbf, EMB, h, EMB, DINNER, 1); // += residual
  }

  rmsnorm256_bf16_kernel<<<NTOK, 256, 0, stream>>>(h, normf_w, hbf);
  convert_pad_kernel<<<(EMB * EMB + 255) / 256, 256, 0, stream>>>(head_w, Wbf, EMB, EMB, EMB);
  dim3 g3(EMB / 64, NTOK / 64);
  gemm_bf16_kernel<<<g3, 256, 0, stream>>>(hbf, EMB, Wbf, EMB, (float*)d_out, EMB, EMB, 0);
}
